// MoEBlock_50242527428752
// MI455X (gfx1250) — matrix operand order repaired and checked by reading, not compiled, not run on hardware
//
#include <hip/hip_runtime.h>
#include <hip/hip_bf16.h>

// Problem constants (from reference): B=2, S=2048, D=1024, E=8, H=4096, K=2
#define NT   4096          // B*S tokens
#define DD   1024
#define HH   4096
#define NE   8
#define TILE_M 32          // tokens per workgroup (2 x 16-row WMMA sub-tiles)
#define HC     128         // H-chunk streamed through LDS per iteration

typedef __attribute__((ext_vector_type(16))) __bf16 v16bf;
typedef __attribute__((ext_vector_type(8)))  float  v8f;
typedef __attribute__((address_space(3)))    __bf16 lds_bf16;

union Frag {
    v16bf v;
    uint4 u4[2];
};

// ---------------------------------------------------------------------------
// A-fragment (16x32 bf16, M x K) from a row-major [16][ld] bf16 buffer in LDS.
// ISA 7.12.2: lanes 0-15 -> M=lane, K pairs 0..7 in VGPR0-3, 16..23 in VGPR4-7;
// lanes 16-31 -> same M, K 8..15 and 24..31.
__device__ inline v16bf load_a_lds(const __bf16* base, int ld, int k0, int lane) {
    int m    = lane & 15;
    int koff = (lane >> 4) << 3;              // 0 or 8
    Frag f;
    const __bf16* p = base + m * ld + k0 + koff;
    f.u4[0] = *(const uint4*)(p);             // K = k0+koff   .. +7
    f.u4[1] = *(const uint4*)(p + 16);        // K = k0+16+koff.. +7
    return f.v;
}

// B-fragment (32x16 bf16, K x N) from a "transposed" global weight Wt[n][k]
// (K contiguous). Lanes 0-15 hold K=0..7,16..23 for N=lane, lanes 16-31 hold K=8..15,24..31.
__device__ inline v16bf load_b_glob(const __bf16* Wt, long ld, int n0, int k0, int lane) {
    int n  = lane & 15;
    int kb = (lane >> 4) << 3;                // 0 or 8
    Frag f;
    const __bf16* p = Wt + (long)(n0 + n) * ld + k0 + kb;
    f.u4[0] = *(const uint4*)(p);
    f.u4[1] = *(const uint4*)(p + 16);
    return f.v;
}

__device__ inline float gelu_f(float x) {
    float x3 = x * x * x;
    return 0.5f * x * (1.0f + tanhf(0.7978845608028654f * (x + 0.044715f * x3)));
}

// ---------------------------------------------------------------------------
// 0) zero output + per-expert counters
__global__ void moe_zero_kernel(float* out, int* counts, long n) {
    long i = (long)blockIdx.x * blockDim.x + threadIdx.x;
    if (i < n) out[i] = 0.0f;
    if (i < NE) counts[i] = 0;
}

// 1a) fp32 -> bf16 weight conversion + transpose for WMMA B-operand reads.
__global__ void moe_convert_w_kernel(const float* __restrict__ w1,
                                     const float* __restrict__ w2,
                                     __bf16* __restrict__ w1t,
                                     __bf16* __restrict__ w2t) {
    long n = (long)NE * DD * HH;
    long i = (long)blockIdx.x * blockDim.x + threadIdx.x;
    if (i >= n) return;
    long e = i / ((long)DD * HH);
    long r = i % ((long)DD * HH);
    {   // w1[e][d][h] -> w1t[e][h][d]
        long d = r / HH, h = r % HH;
        w1t[e * (long)HH * DD + h * DD + d] = (__bf16)w1[i];
    }
    {   // w2[e][h][d] -> w2t[e][d][h]
        long h = r / DD, d = r % DD;
        w2t[e * (long)DD * HH + d * HH + h] = (__bf16)w2[i];
    }
}

// 1b) fp32 -> bf16 activation copy (FFN kernel async-DMAs X rows into LDS).
__global__ void moe_convert_x_kernel(const float* __restrict__ x,
                                     __bf16* __restrict__ xb, long n) {
    long i = (long)blockIdx.x * blockDim.x + threadIdx.x;
    if (i < n) xb[i] = (__bf16)x[i];
}

// 2) router: logits = x @ router_w + b, softmax -> top-2 -> renormalize;
//    builds per-expert token lists (grouped-GEMM style).
__global__ void moe_router_kernel(const float* __restrict__ x,
                                  const float* __restrict__ rw,
                                  const float* __restrict__ rb,
                                  int* __restrict__ counts,
                                  int* __restrict__ tidx,
                                  float* __restrict__ tprob) {
    int wv   = threadIdx.x >> 5;
    int lane = threadIdx.x & 31;
    int tok  = blockIdx.x * 8 + wv;
    if (tok >= NT) return;

    float acc[NE];
#pragma unroll
    for (int e = 0; e < NE; ++e) acc[e] = 0.0f;

    const float* xr = x + (long)tok * DD;
    for (int d = lane; d < DD; d += 32) {
        float xv = xr[d];
#pragma unroll
        for (int e = 0; e < NE; ++e) acc[e] += xv * rw[d * NE + e];
    }
#pragma unroll
    for (int e = 0; e < NE; ++e) {
#pragma unroll
        for (int off = 16; off > 0; off >>= 1)
            acc[e] += __shfl_xor(acc[e], off, 32);
    }
    if (lane == 0) {
        float lg[NE];
        float mx = -1e30f;
#pragma unroll
        for (int e = 0; e < NE; ++e) { lg[e] = acc[e] + rb[e]; mx = fmaxf(mx, lg[e]); }
        int e0 = 0; float v0 = lg[0];
#pragma unroll
        for (int e = 1; e < NE; ++e) if (lg[e] > v0) { v0 = lg[e]; e0 = e; }
        int e1 = -1; float v1 = -1e30f;
#pragma unroll
        for (int e = 0; e < NE; ++e) if (e != e0 && lg[e] > v1) { v1 = lg[e]; e1 = e; }
        float p0 = __expf(v0 - mx), p1 = __expf(v1 - mx);
        float s  = p0 + p1;
        p0 /= s; p1 /= s;
        int pos0 = atomicAdd(&counts[e0], 1);
        tidx[e0 * NT + pos0]  = tok;
        tprob[e0 * NT + pos0] = p0;
        int pos1 = atomicAdd(&counts[e1], 1);
        tidx[e1 * NT + pos1]  = tok;
        tprob[e1 * NT + pos1] = p1;
    }
}

// 3) fused expert FFN: out += p * ( gelu(X @ W1 + b1) @ W2 + b2 )
//    Grid = (E, NT/TILE_M). 32 tokens / workgroup as two 16-row WMMA tiles
//    sharing every B-fragment. B-fragments are register double-buffered so
//    the s_wait for a fragment lands after a full WMMA pair of execution.
__global__ void __launch_bounds__(256)
moe_ffn_kernel(const __bf16* __restrict__ xb,    // [NT][D] bf16
               const __bf16* __restrict__ w1t,   // [E][H][D]
               const float* __restrict__ b1,     // [E][H]
               const __bf16* __restrict__ w2t,   // [E][D][H]
               const float* __restrict__ b2,     // [E][D]
               const int* __restrict__ counts,
               const int* __restrict__ tidx,
               const float* __restrict__ tprob,
               float* __restrict__ out) {
    __shared__ __align__(16) __bf16 xa[TILE_M * DD];     // 64 KB: X tile, bf16
    __shared__ __align__(16) __bf16 hbuf[TILE_M * HC];   //  8 KB: gelu(h) chunk
    __shared__ int   s_tok[TILE_M];
    __shared__ float s_prob[TILE_M];

    const int e    = blockIdx.x;
    const int tile = blockIdx.y;
    const int cnt  = counts[e];
    if (tile * TILE_M >= cnt) return;                    // uniform across block

    const int tid  = threadIdx.x;
    const int wv   = tid >> 5;
    const int lane = tid & 31;

    if (tid < TILE_M) {
        int r = tile * TILE_M + tid;
        if (r < cnt) { s_tok[tid] = tidx[e * NT + r]; s_prob[tid] = tprob[e * NT + r]; }
        else         { s_tok[tid] = 0;                s_prob[tid] = 0.0f; }
    }
    __syncthreads();

    // --- async gather of the X tile (32 rows x 2KB) into LDS, 16B chunks ---
    {
        const int CHUNKS = TILE_M * (DD / 8);            // 16-byte chunks
        for (int i = tid; i < CHUNKS; i += 256) {
            int m = i >> 7;                              // row (128 chunks/row)
            int c = i & 127;
            const __bf16* src = xb + (long)s_tok[m] * DD + c * 8;
            __bf16* dstg = &xa[m * DD + c * 8];
            unsigned int ldsaddr =
                (unsigned int)(unsigned long long)(lds_bf16*)dstg;
            unsigned long long ga = (unsigned long long)src;
            asm volatile("global_load_async_to_lds_b128 %0, %1, off"
                         :: "v"(ldsaddr), "v"(ga) : "memory");
        }
        asm volatile("s_wait_asynccnt 0x0" ::: "memory");
    }
    __syncthreads();

    const __bf16* w1e = w1t + (long)e * HH * DD;
    const __bf16* w2e = w2t + (long)e * DD * HH;

    v8f oacc0[8], oacc1[8];                              // 2 x (16x128) out slices
#pragma unroll
    for (int j = 0; j < 8; ++j) { oacc0[j] = {}; oacc1[j] = {}; }

    const int mrow = (lane >> 4) << 3;                   // C-layout M offset (0/8)
    const int ncol = lane & 15;

    for (int hc = 0; hc < HH; hc += HC) {
        // ---- stage A: h = X @ W1[:, wave's 16 h-cols], both M sub-tiles ----
        v8f hacc0 = {}, hacc1 = {};
        const int nh = hc + wv * 16;
        v16bf bA = load_b_glob(w1e, DD, nh, 0, lane);    // prefetch k0 = 0
#pragma unroll 4
        for (int k0 = 0; k0 < DD - 32; k0 += 32) {
            v16bf bc = bA;
            bA = load_b_glob(w1e, DD, nh, k0 + 32, lane);   // prefetch next
            v16bf a0 = load_a_lds(xa,           DD, k0, lane);
            v16bf a1 = load_a_lds(xa + 16 * DD, DD, k0, lane);
            hacc0 = __builtin_amdgcn_wmma_f32_16x16x32_bf16(
                        false, a0, false, bc, (short)0, hacc0, false, false);
            hacc1 = __builtin_amdgcn_wmma_f32_16x16x32_bf16(
                        false, a1, false, bc, (short)0, hacc1, false, false);
        }
        {   // peeled final k-step (k0 = DD-32), consumes last prefetch
            v16bf a0 = load_a_lds(xa,           DD, DD - 32, lane);
            v16bf a1 = load_a_lds(xa + 16 * DD, DD, DD - 32, lane);
            hacc0 = __builtin_amdgcn_wmma_f32_16x16x32_bf16(
                        false, a0, false, bA, (short)0, hacc0, false, false);
            hacc1 = __builtin_amdgcn_wmma_f32_16x16x32_bf16(
                        false, a1, false, bA, (short)0, hacc1, false, false);
        }
        __syncthreads();                                 // hbuf consumers done
        float bias = b1[e * HH + nh + ncol];
#pragma unroll
        for (int r = 0; r < 8; ++r) {
            float v0g = gelu_f(hacc0[r] + bias);
            float v1g = gelu_f(hacc1[r] + bias);
            hbuf[(r + mrow) * HC + wv * 16 + ncol]      = (__bf16)v0g;
            hbuf[(16 + r + mrow) * HC + wv * 16 + ncol] = (__bf16)v1g;
        }
        __syncthreads();                                 // hbuf ready

        // ---- stage B: oacc += h_chunk @ W2[hc..hc+HC, wave's 128 cols] ----
        v16bf bB = load_b_glob(w2e, HH, wv * 128, hc, lane);  // prefetch (0,0)
#pragma unroll
        for (int kk = 0; kk < HC; kk += 32) {
            v16bf a0 = load_a_lds(hbuf,           HC, kk, lane);
            v16bf a1 = load_a_lds(hbuf + 16 * HC, HC, kk, lane);
#pragma unroll
            for (int j = 0; j < 8; ++j) {
                v16bf bc = bB;
                // prefetch next (j,kk) pair; folds at compile time (fully unrolled)
                if (!(kk == HC - 32 && j == 7)) {
                    int jn = (j + 1) & 7;
                    int kn = (j == 7) ? kk + 32 : kk;
                    bB = load_b_glob(w2e, HH, wv * 128 + jn * 16, hc + kn, lane);
                }
                oacc0[j] = __builtin_amdgcn_wmma_f32_16x16x32_bf16(
                               false, a0, false, bc, (short)0, oacc0[j], false, false);
                oacc1[j] = __builtin_amdgcn_wmma_f32_16x16x32_bf16(
                               false, a1, false, bc, (short)0, oacc1[j], false, false);
            }
        }
    }

    // ---- epilogue: +b2, gate, scatter-accumulate (each token in 2 lists) ----
#pragma unroll
    for (int j = 0; j < 8; ++j) {
        int n = wv * 128 + j * 16 + ncol;
        float bias = b2[e * DD + n];
#pragma unroll
        for (int r = 0; r < 8; ++r) {
            int m0 = r + mrow;
            int m1 = 16 + r + mrow;
            atomicAdd(&out[(long)s_tok[m0] * DD + n], (oacc0[j][r] + bias) * s_prob[m0]);
            atomicAdd(&out[(long)s_tok[m1] * DD + n], (oacc1[j][r] + bias) * s_prob[m1]);
        }
    }
}

// ---------------------------------------------------------------------------
extern "C" void kernel_launch(void* const* d_in, const int* in_sizes, int n_in,
                              void* d_out, int out_size, void* d_ws, size_t ws_size,
                              hipStream_t stream) {
    const float* x   = (const float*)d_in[0];
    const float* rw  = (const float*)d_in[1];
    const float* rb  = (const float*)d_in[2];
    const float* w1  = (const float*)d_in[3];
    const float* b1  = (const float*)d_in[4];
    const float* w2  = (const float*)d_in[5];
    const float* b2  = (const float*)d_in[6];
    float* out = (float*)d_out;

    // workspace layout
    char* ws = (char*)d_ws;
    int*    counts = (int*)ws;                                   // E
    int*    tidx   = (int*)(ws + 256);                           // E*NT
    float*  tprob  = (float*)(ws + 256 + (size_t)NE * NT * 4);   // E*NT
    __bf16* w1t    = (__bf16*)(ws + 256 + (size_t)NE * NT * 8);  // E*H*D
    __bf16* w2t    = w1t + (size_t)NE * HH * DD;                 // E*D*H
    __bf16* xb     = w2t + (size_t)NE * DD * HH;                 // NT*D

    // 0) zero output + counters
    long nzero = (long)NT * DD;
    moe_zero_kernel<<<(nzero + 255) / 256, 256, 0, stream>>>(out, counts, nzero);

    // 1) bf16 conversions (weights transposed; fit in 192MB L2 afterwards)
    long ncvt = (long)NE * DD * HH;
    moe_convert_w_kernel<<<(ncvt + 255) / 256, 256, 0, stream>>>(w1, w2, w1t, w2t);
    long nx = (long)NT * DD;
    moe_convert_x_kernel<<<(nx + 255) / 256, 256, 0, stream>>>(x, xb, nx);

    // 2) router + top-2 + grouped token lists
    moe_router_kernel<<<NT / 8, 256, 0, stream>>>(x, rw, rb, counts, tidx, tprob);

    // 3) fused expert FFN (WMMA bf16, async X-tile staging, double-buffered B)
    dim3 grid(NE, NT / TILE_M);
    moe_ffn_kernel<<<grid, 256, 0, stream>>>(xb, w1t, b1, w2t, b2,
                                             counts, tidx, tprob, out);
}
